// DynamicFilter_50757923504570
// MI455X (gfx1250) — compile-verified, matrix operand order checked by reading, not run on hardware
//
#include <hip/hip_runtime.h>
#include <hip/hip_bf16.h>
#include <math.h>

typedef __attribute__((ext_vector_type(16))) _Float16 v16h;
typedef __attribute__((ext_vector_type(8)))  float    v8f;

#define B_       64
#define T_       4096
#define RNN      1024
#define HYP      256
#define CH       32
#define KS       31
#define KPAD     32
#define PADW     15
#define ODIM     128
#define TCHUNK   512
#define NWAVES   8
#define NTHREADS 256

// ---------------------------------------------------------------------------
// Kernel 1: hypernetwork + filter folding.
//   h  = tanh(q @ W1^T + b1)            (64x256)
//   cw = h @ W2^T + b2                  (64x992, kept in LDS)
//   G[k,o] = sum_c cw[c,k] * Wfc[o,c]   (64x31x128 effective filter, f16)
// One block per batch row; ~83 MFLOP total, pure VALU, negligible runtime.
// G is written k-padded to 32 rows (row 31 = 0) so the hot kernel can use
// K=32 WMMA directly.
// ---------------------------------------------------------------------------
__global__ __launch_bounds__(NTHREADS)
void hypernet_kernel(const float* __restrict__ query,
                     const float* __restrict__ W1, const float* __restrict__ b1,
                     const float* __restrict__ W2, const float* __restrict__ b2,
                     const float* __restrict__ Wfc,
                     _Float16* __restrict__ g_out) {
    __shared__ float s_q[RNN];
    __shared__ float s_h[HYP];
    __shared__ float s_cw[CH * KS];

    const int b   = blockIdx.x;
    const int tid = threadIdx.x;

    for (int i = tid; i < RNN; i += NTHREADS)
        s_q[i] = query[b * RNN + i];
    __syncthreads();

    {   // h: one element per thread (HYP == NTHREADS)
        const float* w = W1 + tid * RNN;
        float acc = 0.f;
        #pragma unroll 8
        for (int k = 0; k < RNN; ++k)
            acc = fmaf(s_q[k], w[k], acc);
        s_h[tid] = tanhf(acc + b1[tid]);
    }
    __syncthreads();

    for (int j = tid; j < CH * KS; j += NTHREADS) {
        const float* w = W2 + j * HYP;
        float acc = 0.f;
        #pragma unroll 8
        for (int k = 0; k < HYP; ++k)
            acc = fmaf(s_h[k], w[k], acc);
        s_cw[j] = acc + b2[j];               // cw[c*KS + k]
    }
    __syncthreads();

    // G[k, o] = sum_c cw[c, k] * Wfc[o, c];  row k==31 zeroed (K padding)
    for (int idx = tid; idx < KPAD * ODIM; idx += NTHREADS) {
        const int k = idx >> 7;              // /128
        const int o = idx & (ODIM - 1);
        float acc = 0.f;
        if (k < KS) {
            #pragma unroll
            for (int c = 0; c < CH; ++c)
                acc = fmaf(s_cw[c * KS + k], Wfc[o * CH + c], acc);
        }
        g_out[(size_t)b * (KPAD * ODIM) + idx] = (_Float16)acc;
    }
}

// ---------------------------------------------------------------------------
// Kernel 2: out[b, t, :] = window(pa, t) @ G[b] + bfc  via WMMA.
// For a 16-t tile:  A = pa sliding window (16x32 f16, K slots straight from
// the halo'ed LDS slice), B = G chunk (32x16 f16, preloaded in registers),
// D = 16x16 f32 accumulated on top of bias.  Zero hot-loop FMAs; the kernel
// is purely bound by the 134 MB output write.
// Grid: (T/TCHUNK, B).  256 threads = 8 wave32.
// ---------------------------------------------------------------------------
__global__ __launch_bounds__(NTHREADS)
void dynconv_wmma_kernel(const float* __restrict__ prev_attn,
                         const _Float16* __restrict__ G,
                         const float* __restrict__ bfc,
                         float* __restrict__ out) {
    __shared__ _Float16 s_pa[TCHUNK + 2 * PADW + 2];   // 544 (K reaches tl+31)
    __shared__ _Float16 s_g[KPAD * ODIM];              // 4096 halves = 8 KB
    __shared__ float    s_bfc[ODIM];

    const int b   = blockIdx.y;
    const int t0  = blockIdx.x * TCHUNK;
    const int tid = threadIdx.x;

    __builtin_prefetch(&prev_attn[(size_t)b * T_ + t0], 0, 3);
    __builtin_prefetch(&G[(size_t)b * (KPAD * ODIM)], 0, 3);

    // ---- stage operands into LDS -----------------------------------------
    for (int i = tid; i < TCHUNK + 2 * PADW + 2; i += NTHREADS) {
        const int g = t0 - PADW + i;
        s_pa[i] = (g >= 0 && g < T_) ? (_Float16)prev_attn[(size_t)b * T_ + g]
                                     : (_Float16)0.f;
    }
    for (int i = tid; i < KPAD * ODIM; i += NTHREADS)
        s_g[i] = G[(size_t)b * (KPAD * ODIM) + i];
    for (int i = tid; i < ODIM; i += NTHREADS)
        s_bfc[i] = bfc[i];
    __syncthreads();

    const int wave  = tid >> 5;
    const int lane  = tid & 31;
    const int half_ = (lane < 16) ? 0 : 8;   // CDNA5 16-bit K-split select
    const int ln16  = lane & 15;

    // ---- B fragments: G as eight 32x16 f16 tiles, built once -------------
    // 16-bit B 32x16 layout: lane = N (mod 16); half h<8 -> K=half_+h,
    // half h>=8 -> K=half_+16+(h-8).
    v16h  bfrag[8];
    float bias[8];
    #pragma unroll
    for (int j = 0; j < 8; ++j) {
        const int o = j * 16 + ln16;
        #pragma unroll
        for (int h = 0; h < 8; ++h) {
            bfrag[j][h]     = s_g[(half_ + h) * ODIM + o];
            bfrag[j][h + 8] = s_g[(half_ + 16 + h) * ODIM + o];
        }
        bias[j] = s_bfc[o];
    }

    // ---- t-tile loop: one WMMA GEMM per 16x128 output tile ---------------
    const int ntiles = TCHUNK / 16;          // 32 tiles, 4 per wave
    for (int tt = wave; tt < ntiles; tt += NWAVES) {
        const int tl0 = tt * 16;
        const int tl  = tl0 + ln16;          // this lane's row M

        // A fragment: sliding window pa[tl + K(h)], same K-slot mapping as B
        v16h afrag;
        #pragma unroll
        for (int h = 0; h < 8; ++h) {
            afrag[h]     = s_pa[tl + half_ + h];
            afrag[h + 8] = s_pa[tl + half_ + 16 + h];
        }

        const int rowbase = tl0 + ((lane < 16) ? 0 : 8);
        float* orow = out + ((size_t)b * T_ + (size_t)(t0 + rowbase)) * ODIM + ln16;
        #pragma unroll
        for (int j = 0; j < 8; ++j) {
            v8f acc;
            #pragma unroll
            for (int r = 0; r < 8; ++r) acc[r] = bias[j];
            acc = __builtin_amdgcn_wmma_f32_16x16x32_f16(
                      /*neg_a=*/false, afrag,
                      /*neg_b=*/false, bfrag[j],
                      /*c_mod=*/(short)0, acc,
                      /*reuse_a=*/false, /*reuse_b=*/false);
            // C/D layout: VGPR r, this lane -> row rowbase + r, col j*16 + ln16
            float* op = orow + j * 16;
            #pragma unroll
            for (int r = 0; r < 8; ++r)
                __builtin_nontemporal_store(acc[r], op + (size_t)r * ODIM);
        }
    }
}

// ---------------------------------------------------------------------------
extern "C" void kernel_launch(void* const* d_in, const int* in_sizes, int n_in,
                              void* d_out, int out_size, void* d_ws, size_t ws_size,
                              hipStream_t stream) {
    const float* query     = (const float*)d_in[0];
    const float* prev_attn = (const float*)d_in[1];
    const float* W1        = (const float*)d_in[2];
    const float* b1        = (const float*)d_in[3];
    const float* W2        = (const float*)d_in[4];
    const float* b2        = (const float*)d_in[5];
    const float* Wfc       = (const float*)d_in[6];
    const float* bfc       = (const float*)d_in[7];
    float*       out       = (float*)d_out;
    _Float16*    g_ws      = (_Float16*)d_ws;      // 64*32*128 f16 = 512 KB

    hypernet_kernel<<<dim3(B_), dim3(NTHREADS), 0, stream>>>(
        query, W1, b1, W2, b2, Wfc, g_ws);

    dynconv_wmma_kernel<<<dim3(T_ / TCHUNK, B_), dim3(NTHREADS), 0, stream>>>(
        prev_attn, g_ws, bfc, out);
}
